// GraphConditioner_43611097924296
// MI455X (gfx1250) — matrix operand order repaired and checked by reading, not compiled, not run on hardware
//
#include <hip/hip_runtime.h>
#include <hip/hip_bf16.h>

typedef __attribute__((ext_vector_type(16))) __bf16 v16bf;
typedef __attribute__((ext_vector_type(8)))  float  v8f;

#define NEG_SLOPE 0.2f
#define GAT_EPS 1e-16f

// ---------- order-preserving float<->uint for atomicMax over signed floats ----------
__device__ __forceinline__ unsigned f2ord(float f) {
    unsigned u = __float_as_uint(f);
    return (u & 0x80000000u) ? ~u : (u | 0x80000000u);
}
__device__ __forceinline__ float ord2f(unsigned u) {
    return __uint_as_float((u & 0x80000000u) ? (u ^ 0x80000000u) : ~u);
}
#define ORD_NEG_INF 0x007FFFFFu   // f2ord(-inf)

// ---------- utility: fill a u32 buffer ----------
__global__ void fill_u32_kernel(unsigned* p, unsigned v, int n) {
    int i = blockIdx.x * blockDim.x + threadIdx.x;
    if (i < n) p[i] = v;
}

// ---------- fused transpose+convert: W[K,256] f32 -> Wt[256,K] bf16 ----------
__global__ void transpose_cvt_kernel(const float* __restrict__ W, __bf16* __restrict__ Wt, int K) {
    int i = blockIdx.x * blockDim.x + threadIdx.x;
    if (i >= K * 256) return;
    int k = i >> 8;          // /256
    int n = i & 255;
    Wt[(size_t)n * K + k] = (__bf16)W[i];
}

// ---------- GEMM: C[M,256] = A[M,K] @ W  (W pre-transposed+bf16: Bt[256,K]) ----------
// One wave computes a 16x64 strip: 1 A-fragment (f32->bf16 in-register) reused across
// 4 B-fragments per k-step; B fragments are direct 32B bf16 vector loads (no cvt).
// v_wmma_f32_16x16x32_bf16, f32 accumulation.
__global__ void gemm_bf16_wmma_kernel(const float* __restrict__ A,
                                      const __bf16* __restrict__ Bt,
                                      float* __restrict__ C,
                                      int M, int K) {
    const int lane = threadIdx.x & 31;
    const int wave = (blockIdx.x * blockDim.x + threadIdx.x) >> 5;
    const int mtiles = M >> 4;                 // M multiple of 16
    if (wave >= mtiles * 4) return;
    const int mt   = wave >> 2;
    const int n0   = (wave & 3) << 6;          // 0,64,128,192
    const int half = lane >> 4;
    const int l16  = lane & 15;

    const float* arow = A + (size_t)(mt * 16 + l16) * K;

    v8f c[4];
    #pragma unroll
    for (int f = 0; f < 4; ++f) {
        v8f z = {0.f, 0.f, 0.f, 0.f, 0.f, 0.f, 0.f, 0.f};
        c[f] = z;
    }

    for (int kb = 0; kb < K; kb += 32) {
        // A fragment (contiguous k order, matches B): elem j -> K=kb+16*half+j
        v16bf a;
        {
            const float* p0 = arow + kb + 16 * half;
            const float* p1 = arow + kb + 16 * half + 8;
            #pragma unroll
            for (int j = 0; j < 8; ++j) a[j] = (__bf16)p0[j];
            #pragma unroll
            for (int j = 0; j < 8; ++j) a[8 + j] = (__bf16)p1[j];
        }
        #pragma unroll
        for (int f = 0; f < 4; ++f) {
            // B fragment: lane holds column n = n0+16f+l16; elem j -> K = kb + 16*half + j
            // 16 contiguous bf16 (32B, 32B-aligned since K%16==0) -> direct vector load.
            const __bf16* bp = Bt + (size_t)(n0 + 16 * f + l16) * K + kb + 16 * half;
            v16bf b = *(const v16bf*)bp;
            c[f] = __builtin_amdgcn_wmma_f32_16x16x32_bf16(
                false, a, false, b, (short)0, c[f], false, false);
        }
    }

    // C layout: VGPR r, lanes0-15 -> M=r, lanes16-31 -> M=8+r; N = l16
    const int mbase = mt * 16 + 8 * half;
    #pragma unroll
    for (int f = 0; f < 4; ++f) {
        const int n = n0 + 16 * f + l16;
        #pragma unroll
        for (int r = 0; r < 8; ++r)
            C[(size_t)(mbase + r) * 256 + n] = c[f][r];
    }
}

// ---------- per-node dual dot: as[n]=h[n]·a_src, ad[n]=h[n]·a_dst (wave per node) ----------
__global__ void node_dots_kernel(const float* __restrict__ H,
                                 const float* __restrict__ asrc,
                                 const float* __restrict__ adst,
                                 float* __restrict__ as_, float* __restrict__ ad_, int N) {
    int wv = (blockIdx.x * blockDim.x + threadIdx.x) >> 5;
    int lane = threadIdx.x & 31;
    if (wv >= N) return;
    const float* row = H + (size_t)wv * 256;
    float s0 = 0.f, s1 = 0.f;
    #pragma unroll
    for (int j = 0; j < 8; ++j) {
        float v = row[lane + 32 * j];
        s0 += v * asrc[lane + 32 * j];
        s1 += v * adst[lane + 32 * j];
    }
    #pragma unroll
    for (int off = 16; off; off >>= 1) {
        s0 += __shfl_xor(s0, off, 32);
        s1 += __shfl_xor(s1, off, 32);
    }
    if (lane == 0) { as_[wv] = s0; ad_[wv] = s1; }
}

// ---------- edge pass 1: e = leaky_relu(as[src]+ad[dst]); segment max by dst ----------
__global__ void edge_pass1_kernel(const int* __restrict__ ei, int E, int N,
                                  const float* __restrict__ as_, const float* __restrict__ ad_,
                                  float* __restrict__ ebuf, unsigned* __restrict__ emax) {
    int i = blockIdx.x * blockDim.x + threadIdx.x;
    int EE = E + N;
    if (i >= EE) return;
    int s, d;
    if (i < E) { s = ei[i]; d = ei[E + i]; } else { s = d = i - E; }
    float e = as_[s] + ad_[d];
    e = (e > 0.f) ? e : NEG_SLOPE * e;
    ebuf[i] = e;
    atomicMax(emax + d, f2ord(e));
}

// ---------- edge pass 2: w = exp(e - max[dst]); segment sum -> denom ----------
__global__ void edge_pass2_kernel(const int* __restrict__ ei, int E, int N,
                                  float* __restrict__ ebuf, const unsigned* __restrict__ emax,
                                  float* __restrict__ denom) {
    int i = blockIdx.x * blockDim.x + threadIdx.x;
    int EE = E + N;
    if (i >= EE) return;
    int d = (i < E) ? ei[E + i] : (i - E);
    float w = __expf(ebuf[i] - ord2f(emax[d]));
    ebuf[i] = w;
    atomicAdd(denom + d, w);
}

// ---------- edge pass 3: acc[dst] += (w/(denom[dst]+eps)) * P[src]  (wave per edge) ----------
__global__ void edge_scatter_kernel(const int* __restrict__ ei, int E, int N,
                                    const float* __restrict__ ebuf, const float* __restrict__ denom,
                                    const float* __restrict__ P, float* __restrict__ ACC) {
    int t = blockIdx.x * blockDim.x + threadIdx.x;
    int i = t >> 5;
    int lane = t & 31;
    int EE = E + N;
    if (i >= EE) return;
    int s, d;
    if (i < E) { s = ei[i]; d = ei[E + i]; } else { s = d = i - E; }
    float alpha = ebuf[i] / (denom[d] + GAT_EPS);
    const float4* hp = (const float4*)(P + (size_t)s * 256);
    float4 v0 = hp[lane];
    float4 v1 = hp[lane + 32];
    float* ab = ACC + (size_t)d * 256;
    int c0 = lane * 4;
    atomicAdd(ab + c0 + 0, alpha * v0.x);
    atomicAdd(ab + c0 + 1, alpha * v0.y);
    atomicAdd(ab + c0 + 2, alpha * v0.z);
    atomicAdd(ab + c0 + 3, alpha * v0.w);
    atomicAdd(ab + 128 + c0 + 0, alpha * v1.x);
    atomicAdd(ab + 128 + c0 + 1, alpha * v1.y);
    atomicAdd(ab + 128 + c0 + 2, alpha * v1.z);
    atomicAdd(ab + 128 + c0 + 3, alpha * v1.w);
}

// ---------- H = relu(ACC + bias) ----------
__global__ void finalize_layer_kernel(const float* __restrict__ ACC, const float* __restrict__ bias,
                                      float* __restrict__ H, int total) {
    int i = blockIdx.x * blockDim.x + threadIdx.x;
    if (i >= total) return;
    float v = ACC[i] + bias[i & 255];
    H[i] = v > 0.f ? v : 0.f;
}

// ---------- gate[n] = h[n]·gate_w + gate_b ; track global max (wave per node) ----------
__global__ void node_gate_kernel(const float* __restrict__ H, const float* __restrict__ gw,
                                 const float* __restrict__ gb, float* __restrict__ gate,
                                 unsigned* __restrict__ gmax, int N) {
    int wv = (blockIdx.x * blockDim.x + threadIdx.x) >> 5;
    int lane = threadIdx.x & 31;
    if (wv >= N) return;
    const float* row = H + (size_t)wv * 256;
    float s = 0.f;
    #pragma unroll
    for (int j = 0; j < 8; ++j) s += row[lane + 32 * j] * gw[lane + 32 * j];
    #pragma unroll
    for (int off = 16; off; off >>= 1) s += __shfl_xor(s, off, 32);
    if (lane == 0) {
        float g = s + gb[0];
        gate[wv] = g;
        atomicMax(gmax, f2ord(g));
    }
}

// ---------- gate[n] = exp(gate[n]-gmax); gsum += ----------
__global__ void pool_exp_kernel(float* __restrict__ gate, const unsigned* __restrict__ gmax,
                                float* __restrict__ gsum, int N) {
    int i = blockIdx.x * blockDim.x + threadIdx.x;
    if (i >= N) return;
    float w = __expf(gate[i] - ord2f(*gmax));
    gate[i] = w;
    atomicAdd(gsum, w);
}

// ---------- pacc[f] += sum_{n in chunk} gate[n]*H[n,f]  (block=256 threads, 128 nodes/block) ----------
__global__ void pool_wsum_kernel(const float* __restrict__ gate, const float* __restrict__ H,
                                 float* __restrict__ pacc, int N) {
    int f = threadIdx.x;
    int n0 = blockIdx.x * 128;
    int n1 = n0 + 128; if (n1 > N) n1 = N;
    float s = 0.f;
    for (int n = n0; n < n1; ++n) s += gate[n] * H[(size_t)n * 256 + f];
    atomicAdd(pacc + f, s);
}

__global__ void pool_final_kernel(const float* __restrict__ pacc, const float* __restrict__ gsum,
                                  float* __restrict__ out) {
    int f = threadIdx.x;
    out[f] = pacc[f] / (*gsum);
}

// =====================================================================================
extern "C" void kernel_launch(void* const* d_in, const int* in_sizes, int n_in,
                              void* d_out, int out_size, void* d_ws, size_t ws_size,
                              hipStream_t stream) {
    const float* x       = (const float*)d_in[0];
    const int*   ei      = (const int*)d_in[1];
    const float* W0      = (const float*)d_in[2];
    const float* a_src0  = (const float*)d_in[3];
    const float* a_dst0  = (const float*)d_in[4];
    const float* b0      = (const float*)d_in[5];
    const float* W1      = (const float*)d_in[6];
    const float* a_src1  = (const float*)d_in[7];
    const float* a_dst1  = (const float*)d_in[8];
    const float* b1      = (const float*)d_in[9];
    const float* gate_w  = (const float*)d_in[10];
    const float* gate_b  = (const float*)d_in[11];

    const int N  = in_sizes[0] / 768;
    const int E  = in_sizes[1] / 2;
    const int EE = E + N;

    // ---- workspace carve-out ----
    char* ws = (char*)d_ws;
    size_t off = 0;
    auto carve = [&](size_t bytes) {
        void* p = ws + off;
        off = (off + bytes + 255) & ~(size_t)255;
        return p;
    };
    __bf16*   W0t   = (__bf16*)carve((size_t)768 * 256 * 2);
    __bf16*   W1t   = (__bf16*)carve((size_t)256 * 256 * 2);
    float*    P     = (float*)carve((size_t)N * 256 * 4);   // projection h = A@W
    float*    ACC   = (float*)carve((size_t)N * 256 * 4);   // aggregation accumulator
    float*    H     = (float*)carve((size_t)N * 256 * 4);   // layer output (post relu)
    float*    as_   = (float*)carve((size_t)N * 4);
    float*    ad_   = (float*)carve((size_t)N * 4);
    unsigned* emax  = (unsigned*)carve((size_t)N * 4);
    float*    denom = (float*)carve((size_t)N * 4);
    float*    ebuf  = (float*)carve((size_t)EE * 4);
    float*    gate  = (float*)carve((size_t)N * 4);
    unsigned* gmaxp = (unsigned*)carve(256);
    float*    gsump = (float*)carve(256);
    float*    pacc  = (float*)carve((size_t)256 * 4);

    const int TB = 256;
    auto blocks = [](long long n, int tb) { return (int)((n + tb - 1) / tb); };

    // ---- weight transpose + bf16 convert (once per launch; deterministic) ----
    transpose_cvt_kernel<<<blocks(768 * 256, TB), TB, 0, stream>>>(W0, W0t, 768);
    transpose_cvt_kernel<<<blocks(256 * 256, TB), TB, 0, stream>>>(W1, W1t, 256);

    const int gemm_waves  = (N / 16) * 4;
    const int gemm_blocks = blocks((long long)gemm_waves * 32, TB);

    for (int layer = 0; layer < 2; ++layer) {
        const float*  A    = (layer == 0) ? x : H;
        const __bf16* Bt   = (layer == 0) ? W0t : W1t;
        const int     K    = (layer == 0) ? 768 : 256;
        const float*  asrc = (layer == 0) ? a_src0 : a_src1;
        const float*  adst = (layer == 0) ? a_dst0 : a_dst1;
        const float*  bias = (layer == 0) ? b0 : b1;

        // projection (WMMA)
        gemm_bf16_wmma_kernel<<<gemm_blocks, TB, 0, stream>>>(A, Bt, P, N, K);
        // attention logits per node
        node_dots_kernel<<<blocks((long long)N * 32, TB), TB, 0, stream>>>(P, asrc, adst, as_, ad_, N);
        // reset segment buffers
        fill_u32_kernel<<<blocks(N, TB), TB, 0, stream>>>(emax, ORD_NEG_INF, N);
        fill_u32_kernel<<<blocks(N, TB), TB, 0, stream>>>((unsigned*)denom, 0u, N);
        fill_u32_kernel<<<blocks((long long)N * 256, TB), TB, 0, stream>>>((unsigned*)ACC, 0u, N * 256);
        // edge softmax + scatter
        edge_pass1_kernel<<<blocks(EE, TB), TB, 0, stream>>>(ei, E, N, as_, ad_, ebuf, emax);
        edge_pass2_kernel<<<blocks(EE, TB), TB, 0, stream>>>(ei, E, N, ebuf, emax, denom);
        edge_scatter_kernel<<<blocks((long long)EE * 32, TB), TB, 0, stream>>>(ei, E, N, ebuf, denom, P, ACC);
        // bias + relu
        finalize_layer_kernel<<<blocks((long long)N * 256, TB), TB, 0, stream>>>(ACC, bias, H, N * 256);
    }

    // ---- global attention pooling ----
    fill_u32_kernel<<<1, 1, 0, stream>>>(gmaxp, ORD_NEG_INF, 1);
    fill_u32_kernel<<<1, 1, 0, stream>>>((unsigned*)gsump, 0u, 1);
    fill_u32_kernel<<<1, TB, 0, stream>>>((unsigned*)pacc, 0u, 256);

    node_gate_kernel<<<blocks((long long)N * 32, TB), TB, 0, stream>>>(H, gate_w, gate_b, gate, gmaxp, N);
    pool_exp_kernel<<<blocks(N, TB), TB, 0, stream>>>(gate, gmaxp, gsump, N);
    pool_wsum_kernel<<<blocks(N, 128), TB, 0, stream>>>(gate, H, pacc, N);
    pool_final_kernel<<<1, 256, 0, stream>>>(pacc, gsump, (float*)d_out);
}
